// BiLSTM_Attention_19653770347058
// MI455X (gfx1250) — compile-verified, matrix operand order checked by reading, not run on hardware
//
#include <hip/hip_runtime.h>

// ---------------------------------------------------------------------------
// CDNA5 (gfx1250) WMMA types & helpers
// ---------------------------------------------------------------------------
typedef _Float16 v16h __attribute__((ext_vector_type(16)));
typedef _Float16 v8h  __attribute__((ext_vector_type(8)));
typedef float    v8f  __attribute__((ext_vector_type(8)));

#define GAT_ALPHA 0.2f
#define GAT_MU    0.001f

__device__ __forceinline__ v8f zero8f() {
  v8f z;
#pragma unroll
  for (int i = 0; i < 8; ++i) z[i] = 0.f;
  return z;
}

__device__ __forceinline__ v16h zero16h() {
  v16h z;
#pragma unroll
  for (int i = 0; i < 16; ++i) z[i] = (_Float16)0.f;
  return z;
}

// 16-bit A/B fragment for V_WMMA_F32_16X16X32_F16 (wave32).
// ISA layout (CDNA5 ISA): lane holds row (lane&15); lanes 0-15
// carry K {0..7,16..23}, lanes 16-31 carry K {8..15,24..31} -> two 16B
// contiguous segments per lane from a row-major [row, K] buffer.
__device__ __forceinline__ v16h frag_ld(const _Float16* rowk, int lane) {
  const int hh = (lane >> 4) & 1;
  v8h s0 = *(const v8h*)(rowk + hh * 8);
  v8h s1 = *(const v8h*)(rowk + 16 + hh * 8);
  v16h r;
#pragma unroll
  for (int i = 0; i < 8; ++i) { r[i] = s0[i]; r[i + 8] = s1[i]; }
  return r;
}

__device__ __forceinline__ v8f wmma16(v16h a, v16h b, v8f c) {
  // D = A(16x32 f16) * B(32x16 f16) + C(16x16 f32)
  return __builtin_amdgcn_wmma_f32_16x16x32_f16(
      false, a, false, b, (short)0, c, false, false);
}

__device__ __forceinline__ float sigf(float x) {
  return 1.f / (1.f + __expf(-x));
}

// ---------------------------------------------------------------------------
// Prep: f32 -> f16 weight conversion with zero K/N padding
// ---------------------------------------------------------------------------
// dst[rows, kpad] <- src[rows, kin], zero pad K
__global__ void pad_convert(_Float16* __restrict__ dst,
                            const float* __restrict__ src,
                            int rows, int kin, int kpad) {
  int idx = blockIdx.x * blockDim.x + threadIdx.x;
  if (idx >= rows * kpad) return;
  int r = idx / kpad, k = idx % kpad;
  dst[idx] = (_Float16)(k < kin ? src[r * kin + k] : 0.f);
}

// dst[npad, kpad] <- transpose(src[kin, nin]) with zero pad (B operand for
// h = inp @ W : B[n, k] = W[k, n])
__global__ void transpose_pad(_Float16* __restrict__ dst,
                              const float* __restrict__ src,
                              int nin, int kin, int npad, int kpad) {
  int idx = blockIdx.x * blockDim.x + threadIdx.x;
  if (idx >= npad * kpad) return;
  int n = idx / kpad, k = idx % kpad;
  dst[idx] = (_Float16)((n < nin && k < kin) ? src[k * nin + n] : 0.f);
}

// ---------------------------------------------------------------------------
// BiLSTM: 16 sequences per block, 8 waves (waves 0-3 fwd, 4-7 bwd).
// Gates g[16,400] = x_t[16,128] @ Wih^T + h[16,128] @ Whh^T via WMMA.
// ---------------------------------------------------------------------------
__global__ __launch_bounds__(256) void bilstm_kernel(
    const int* __restrict__ bh, const int* __restrict__ br,
    const int* __restrict__ bt,
    const float* __restrict__ ent, const float* __restrict__ rel,
    const _Float16* __restrict__ WihF, const _Float16* __restrict__ WhhF,
    const float* __restrict__ bF,
    const _Float16* __restrict__ WihB, const _Float16* __restrict__ WhhB,
    const float* __restrict__ bB,
    _Float16* __restrict__ out16, float* __restrict__ out0) {
  __shared__ _Float16 xs[3][16][128];   // gathered inputs (f16, K-padded)
  __shared__ _Float16 hs[2][16][128];   // hidden state per direction
  __shared__ float    cs[2][16][100];   // cell state per direction
  extern __shared__ float gs[];         // [2][16][400] gate pre-activations

  const int tid = threadIdx.x;
  const int m0  = blockIdx.x * 16;

  // gather x (embedding -> f16, zero pad K 100..127)
  for (int idx = tid; idx < 16 * 3 * 128; idx += 256) {
    int row = idx / 384, rem = idx % 384;
    int t = rem >> 7, k = rem & 127;
    int m = m0 + row;
    float v = 0.f;
    if (k < 100) {
      if (t == 0)      v = ent[bh[m] * 100 + k];
      else if (t == 1) v = rel[br[m] * 100 + k];
      else             v = ent[bt[m] * 100 + k];
    }
    xs[t][row][k] = (_Float16)v;
  }
  for (int idx = tid; idx < 2 * 16 * 128; idx += 256)
    ((_Float16*)hs)[idx] = (_Float16)0.f;
  for (int idx = tid; idx < 2 * 16 * 100; idx += 256)
    ((float*)cs)[idx] = 0.f;
  // zero pad cols 600..607 of out16 for these rows
  for (int idx = tid; idx < 16 * 8; idx += 256) {
    int row = idx >> 3, k = 600 + (idx & 7);
    out16[(size_t)(m0 + row) * 608 + k] = (_Float16)0.f;
  }
  __syncthreads();

  const int wave = tid >> 5, lane = tid & 31;
  const int dir = wave >> 2, dwave = wave & 3;   // dir: 0=fwd, 1=bwd
  const _Float16* Wih = dir ? WihB : WihF;
  const _Float16* Whh = dir ? WhhB : WhhF;
  const int r = lane & 15;

  for (int s = 0; s < 3; ++s) {
    const int tx = dir ? (2 - s) : s;            // input timestep
    // ---- GEMM: each dir-wave covers gate tiles nt, nt+4, ... (25 tiles)
    for (int nt = dwave; nt < 25; nt += 4) {
      // speculative prefetch of the next gate tile's weight rows
      const int ntn = (nt + 4 < 25) ? nt + 4 : nt;
      __builtin_prefetch(Wih + (size_t)(ntn * 16 + r) * 128, 0, 0);
      __builtin_prefetch(Whh + (size_t)(ntn * 16 + r) * 128, 0, 0);
      v8f acc = zero8f();
#pragma unroll
      for (int kc = 0; kc < 4; ++kc) {
        v16h a = frag_ld(&xs[tx][r][kc * 32], lane);
        v16h b = frag_ld(Wih + (size_t)(nt * 16 + r) * 128 + kc * 32, lane);
        acc = wmma16(a, b, acc);
      }
#pragma unroll
      for (int kc = 0; kc < 4; ++kc) {
        v16h a = frag_ld(&hs[dir][r][kc * 32], lane);
        v16h b = frag_ld(Whh + (size_t)(nt * 16 + r) * 128 + kc * 32, lane);
        acc = wmma16(a, b, acc);
      }
      const int col = nt * 16 + (lane & 15);
      const int rb  = (lane >> 4) * 8;           // C/D: M = v + 8*(lane>=16)
      float* gd = gs + dir * (16 * 400);
#pragma unroll
      for (int v = 0; v < 8; ++v) gd[(rb + v) * 400 + col] = acc[v];
    }
    __syncthreads();
    // ---- elementwise gates (threads 0-127: fwd, 128-255: bwd)
    {
      const int d  = tid >> 7;
      const int lt = tid & 127;
      const float* bd = d ? bB : bF;
      float* gd = gs + d * (16 * 400);
      for (int idx = lt; idx < 1600; idx += 128) {
        int row = idx / 100, j = idx % 100;
        float gi = gd[row * 400 + j]       + bd[j];
        float gf = gd[row * 400 + 100 + j] + bd[100 + j];
        float gg = gd[row * 400 + 200 + j] + bd[200 + j];
        float go = gd[row * 400 + 300 + j] + bd[300 + j];
        float cc = sigf(gf) * cs[d][row][j] + sigf(gi) * tanhf(gg);
        float hv = sigf(go) * tanhf(cc);
        cs[d][row][j] = cc;
        hs[d][row][j] = (_Float16)hv;
        int tout = d ? (2 - s) : s;              // bwd hidden maps to t = 2-s
        int col = tout * 200 + d * 100 + j;      // out = [of_t | ob_t] per t
        int m = m0 + row;
        out16[(size_t)m * 608 + col] = (_Float16)hv;
        if ((m % 80) == 0) out0[(size_t)(m / 80) * 600 + col] = hv;
      }
    }
    __syncthreads();
  }
}

// ---------------------------------------------------------------------------
// Process LSTM (unidirectional, keep only last hidden state)
// ---------------------------------------------------------------------------
__global__ __launch_bounds__(256) void plstm_kernel(
    const int* __restrict__ bh, const int* __restrict__ br,
    const int* __restrict__ bt,
    const float* __restrict__ ent, const float* __restrict__ rel,
    const _Float16* __restrict__ Wih, const _Float16* __restrict__ Whh,
    const float* __restrict__ bP,
    _Float16* __restrict__ op16, float* __restrict__ out2) {
  __shared__ _Float16 xs[3][16][128];
  __shared__ _Float16 hs[16][128];
  __shared__ float    cs[16][100];
  extern __shared__ float gs[];  // [16][400]

  const int tid = threadIdx.x;
  const int m0  = blockIdx.x * 16;

  for (int idx = tid; idx < 16 * 3 * 128; idx += 256) {
    int row = idx / 384, rem = idx % 384;
    int t = rem >> 7, k = rem & 127;
    int m = m0 + row;
    float v = 0.f;
    if (k < 100) {
      if (t == 0)      v = ent[bh[m] * 100 + k];
      else if (t == 1) v = rel[br[m] * 100 + k];
      else             v = ent[bt[m] * 100 + k];
    }
    xs[t][row][k] = (_Float16)v;
  }
  for (int idx = tid; idx < 16 * 128; idx += 256)
    ((_Float16*)hs)[idx] = (_Float16)0.f;
  for (int idx = tid; idx < 16 * 100; idx += 256)
    ((float*)cs)[idx] = 0.f;
  for (int idx = tid; idx < 16 * 28; idx += 256) {   // zero pad cols 100..127
    int row = idx / 28, k = 100 + idx % 28;
    op16[(size_t)(m0 + row) * 128 + k] = (_Float16)0.f;
  }
  __syncthreads();

  const int wave = tid >> 5, lane = tid & 31;
  const int r = lane & 15;

  for (int s = 0; s < 3; ++s) {
    for (int nt = wave; nt < 25; nt += 8) {
      const int ntn = (nt + 8 < 25) ? nt + 8 : nt;
      __builtin_prefetch(Wih + (size_t)(ntn * 16 + r) * 128, 0, 0);
      __builtin_prefetch(Whh + (size_t)(ntn * 16 + r) * 128, 0, 0);
      v8f acc = zero8f();
#pragma unroll
      for (int kc = 0; kc < 4; ++kc) {
        v16h a = frag_ld(&xs[s][r][kc * 32], lane);
        v16h b = frag_ld(Wih + (size_t)(nt * 16 + r) * 128 + kc * 32, lane);
        acc = wmma16(a, b, acc);
      }
#pragma unroll
      for (int kc = 0; kc < 4; ++kc) {
        v16h a = frag_ld(&hs[r][kc * 32], lane);
        v16h b = frag_ld(Whh + (size_t)(nt * 16 + r) * 128 + kc * 32, lane);
        acc = wmma16(a, b, acc);
      }
      const int col = nt * 16 + (lane & 15);
      const int rb  = (lane >> 4) * 8;
#pragma unroll
      for (int v = 0; v < 8; ++v) gs[(rb + v) * 400 + col] = acc[v];
    }
    __syncthreads();
    for (int idx = tid; idx < 1600; idx += 256) {
      int row = idx / 100, j = idx % 100;
      float gi = gs[row * 400 + j]       + bP[j];
      float gf = gs[row * 400 + 100 + j] + bP[100 + j];
      float gg = gs[row * 400 + 200 + j] + bP[200 + j];
      float go = gs[row * 400 + 300 + j] + bP[300 + j];
      float cc = sigf(gf) * cs[row][j] + sigf(gi) * tanhf(gg);
      float hv = sigf(go) * tanhf(cc);
      cs[row][j] = cc;
      hs[row][j] = (_Float16)hv;
      if (s == 2) {
        int m = m0 + row;
        op16[(size_t)m * 128 + j] = (_Float16)hv;
        if ((m % 80) == 0) out2[(size_t)(m / 80) * 100 + j] = hv;
      }
    }
    __syncthreads();
  }
}

// ---------------------------------------------------------------------------
// GAT: one block per group of 40 nodes. h = inp @ W (WMMA, f32 in LDS),
// e[n] = lrelu(h[n].a_lo + h[0].a_hi), softmax, relu(att-mu), weighted sum.
// Compile-time F/NTILES/KCN/LDK so the K-loop unrolls & pipelines.
// ---------------------------------------------------------------------------
template <int F, int NTILES, int KCN, int LDK>
__global__ __launch_bounds__(256) void gat_kernel(
    const _Float16* __restrict__ inp16,
    const _Float16* __restrict__ Wt,       // [NTILES*16, LDK] = W^T padded
    const float* __restrict__ avec,        // [2F]
    float* __restrict__ dst) {
  extern __shared__ float hsm[];           // [40][NTILES*16]
  __shared__ float ebuf[40], attbuf[40];

  const int g = blockIdx.x;
  const int tid = threadIdx.x, wave = tid >> 5, lane = tid & 31;
  const int r = lane & 15;
  const int stride = NTILES * 16;
  const _Float16* Ab = inp16 + (size_t)g * 40 * LDK;

  const int jobs = 3 * NTILES;             // 3 M-tiles (rows 0..47, 40 valid)
  for (int job = wave; job < jobs; job += 8) {
    int mt = job / NTILES, nt = job % NTILES;
    // speculative prefetch of next job's B tile rows
    int ntn = (job + 8 < jobs) ? (job + 8) % NTILES : nt;
    __builtin_prefetch(Wt + (size_t)(ntn * 16 + r) * LDK, 0, 0);
    int arow = mt * 16 + r;
    bool valid = arow < 40;
    v8f acc = zero8f();
#pragma unroll 4
    for (int kc = 0; kc < KCN; ++kc) {
      v16h a;
      if (valid) a = frag_ld(Ab + (size_t)arow * LDK + kc * 32, lane);
      else       a = zero16h();
      v16h b = frag_ld(Wt + (size_t)(nt * 16 + r) * LDK + kc * 32, lane);
      acc = wmma16(a, b, acc);
    }
    int col = nt * 16 + (lane & 15);
    int rb  = mt * 16 + (lane >> 4) * 8;
#pragma unroll
    for (int v = 0; v < 8; ++v) {
      int row = rb + v;
      if (row < 40 && col < F) hsm[row * stride + col] = acc[v];
    }
  }
  __syncthreads();

  // attention logits: one wave per node, lanes stride features, shfl reduce
  for (int n = wave; n < 40; n += 8) {
    float p = 0.f;
    for (int f = lane; f < F; f += 32)
      p += hsm[n * stride + f] * avec[f] + hsm[f] * avec[F + f];
#pragma unroll
    for (int off = 16; off > 0; off >>= 1)
      p += __shfl_xor(p, off, 32);
    if (lane == 0) ebuf[n] = (p >= 0.f) ? p : GAT_ALPHA * p;
  }
  __syncthreads();
  if (tid == 0) {
    float mx = -3.4e38f;
    for (int n = 0; n < 40; ++n) mx = fmaxf(mx, ebuf[n]);
    float s = 0.f;
    for (int n = 0; n < 40; ++n) {
      float ex = __expf(ebuf[n] - mx);
      attbuf[n] = ex; s += ex;
    }
    float inv = 1.f / s;
    for (int n = 0; n < 40; ++n)
      attbuf[n] = fmaxf(attbuf[n] * inv - GAT_MU, 0.f);
  }
  __syncthreads();
  for (int f = tid; f < F; f += 256) {
    float acc = 0.f;
#pragma unroll 8
    for (int n = 0; n < 40; ++n) acc += attbuf[n] * hsm[n * stride + f];
    dst[(size_t)g * F + f] = acc;
  }
}

// ---------------------------------------------------------------------------
// Host launcher
// ---------------------------------------------------------------------------
extern "C" void kernel_launch(void* const* d_in, const int* in_sizes, int n_in,
                              void* d_out, int out_size, void* d_ws,
                              size_t ws_size, hipStream_t stream) {
  (void)in_sizes; (void)n_in; (void)out_size; (void)ws_size;

  const int*   bh    = (const int*)d_in[0];
  const int*   br    = (const int*)d_in[1];
  const int*   bt    = (const int*)d_in[2];
  const float* ent   = (const float*)d_in[3];
  const float* rel   = (const float*)d_in[4];
  const float* ent1  = (const float*)d_in[5];
  const float* rel1  = (const float*)d_in[6];
  const float* Wih_f = (const float*)d_in[7];
  const float* Whh_f = (const float*)d_in[8];
  const float* b_f   = (const float*)d_in[9];
  const float* Wih_b = (const float*)d_in[10];
  const float* Whh_b = (const float*)d_in[11];
  const float* b_b   = (const float*)d_in[12];
  const float* Wih_p = (const float*)d_in[13];
  const float* Whh_p = (const float*)d_in[14];
  const float* b_p   = (const float*)d_in[15];
  const float* W1    = (const float*)d_in[16];
  const float* a1    = (const float*)d_in[17];
  const float* W2    = (const float*)d_in[18];
  const float* a2    = (const float*)d_in[19];

  float* out  = (float*)d_out;
  float* sec1 = out + 1024 * 600;          // out_att     [2048,600]
  float* sec2 = sec1 + 2048 * 600;         // op2[:,0,:]  [1024,100]
  float* sec3 = sec2 + 1024 * 100;         // output_att  [2048,100]

  char* ws = (char*)d_ws;
  size_t off = 0;
  _Float16* W1t   = (_Float16*)(ws + off); off += (size_t)608 * 608 * 2;
  _Float16* W2t   = (_Float16*)(ws + off); off += (size_t)112 * 128 * 2;
  _Float16* pWihF = (_Float16*)(ws + off); off += (size_t)400 * 128 * 2;
  _Float16* pWhhF = (_Float16*)(ws + off); off += (size_t)400 * 128 * 2;
  _Float16* pWihB = (_Float16*)(ws + off); off += (size_t)400 * 128 * 2;
  _Float16* pWhhB = (_Float16*)(ws + off); off += (size_t)400 * 128 * 2;
  _Float16* pWihP = (_Float16*)(ws + off); off += (size_t)400 * 128 * 2;
  _Float16* pWhhP = (_Float16*)(ws + off); off += (size_t)400 * 128 * 2;
  _Float16* out16 = (_Float16*)(ws + off); off += (size_t)81920 * 608 * 2;
  _Float16* op16  = (_Float16*)(ws + off); off += (size_t)81920 * 128 * 2;

  // weight prep (f32 -> padded f16)
  transpose_pad<<<(608 * 608) / 256, 256, 0, stream>>>(W1t, W1, 600, 600, 608, 608);
  transpose_pad<<<(112 * 128) / 256, 256, 0, stream>>>(W2t, W2, 100, 100, 112, 128);
  pad_convert<<<200, 256, 0, stream>>>(pWihF, Wih_f, 400, 100, 128);
  pad_convert<<<200, 256, 0, stream>>>(pWhhF, Whh_f, 400, 100, 128);
  pad_convert<<<200, 256, 0, stream>>>(pWihB, Wih_b, 400, 100, 128);
  pad_convert<<<200, 256, 0, stream>>>(pWhhB, Whh_b, 400, 100, 128);
  pad_convert<<<200, 256, 0, stream>>>(pWihP, Wih_p, 400, 100, 128);
  pad_convert<<<200, 256, 0, stream>>>(pWhhP, Whh_p, 400, 100, 128);

  // LSTMs (16 sequences per block)
  bilstm_kernel<<<81920 / 16, 256, 2 * 16 * 400 * sizeof(float), stream>>>(
      bh, br, bt, ent, rel, pWihF, pWhhF, b_f, pWihB, pWhhB, b_b, out16, out);
  plstm_kernel<<<81920 / 16, 256, 16 * 400 * sizeof(float), stream>>>(
      bh, br, bt, ent1, rel1, pWihP, pWhhP, b_p, op16, sec2);

  // GAT layers (one block per group of 40)
  gat_kernel<600, 38, 19, 608><<<2048, 256, 40 * 608 * sizeof(float), stream>>>(
      out16, W1t, a1, sec1);
  gat_kernel<100, 7, 4, 128><<<2048, 256, 40 * 112 * sizeof(float), stream>>>(
      op16, W2t, a2, sec3);
}
